// SymGNN_20486994002062
// MI455X (gfx1250) — compile-verified, matrix operand order checked by reading, not run on hardware
//
#include <hip/hip_runtime.h>
#include <cmath>

// ---------------------------------------------------------------------------
// SymGNN forward for MI455X (gfx1250, wave32, WMMA).
// All dense GEMMs go through v_wmma_f32_16x16x32_f16 (f16 in, f32 accum),
// with the weight matrix staged in LDS as f16 (one cooperative convert per
// block), B fragments served by ds_load_b128, A fragments converted inline.
// ---------------------------------------------------------------------------

typedef _Float16 vh16 __attribute__((ext_vector_type(16)));
typedef _Float16 vh8  __attribute__((ext_vector_type(8)));
typedef float    vf16 __attribute__((ext_vector_type(16)));
typedef float    vf8  __attribute__((ext_vector_type(8)));

#define DD 64   // feature dim

__device__ __forceinline__ vh16 load_h16(const float* __restrict__ p) {
    vf16 v = *reinterpret_cast<const vf16*>(p);
    return __builtin_convertvector(v, vh16);
}

__device__ __forceinline__ float atomicMaxF(float* addr, float v) {
    unsigned int* ua = (unsigned int*)addr;
    unsigned int old = *ua;
    while (__uint_as_float(old) < v) {
        unsigned int assumed = old;
        old = atomicCAS(ua, assumed, __float_as_uint(v));
        if (old == assumed) break;
    }
    return __uint_as_float(old);
}

// ---------------------------------------------------------------------------
// C[M x Nout] = X[M x K] @ W[Nout x K]^T (+bias) (+res).  M%16==0, K%32==0,
// Nout%64==0.  One wave => 16 rows x 64 cols (4 WMMA accumulators).
// W is staged f32->f16 into LDS once per block; row stride padded by 8 halves
// (16B) to stagger banks for the lane-strided B-fragment reads.
// kbits = log2(K).  Dynamic LDS bytes = Nout*(K+8)*2.
// ---------------------------------------------------------------------------
__global__ void gemm_nt_wmma(const float* __restrict__ X,
                             const float* __restrict__ W,
                             const float* __restrict__ bias,
                             const float* __restrict__ res,
                             float* __restrict__ C,
                             int M, int K, int Nout, int kbits) {
    extern __shared__ _Float16 lw[];
    const int KP = K + 8;

    // ---- cooperative stage of W (Nout x K) into LDS as f16 ----
    const int total = Nout << kbits;            // Nout * K
    for (int i = threadIdx.x * 8; i < total; i += blockDim.x * 8) {
        const int row = i >> kbits;
        const int k   = i & (K - 1);
        vf8 v = *reinterpret_cast<const vf8*>(W + (long)row * K + k);
        *reinterpret_cast<vh8*>(lw + row * KP + k) = __builtin_convertvector(v, vh8);
    }
    __syncthreads();

    const int  lane    = threadIdx.x & 31;
    const long wid     = ((long)blockIdx.x * blockDim.x + threadIdx.x) >> 5;
    const int  ngroups = Nout >> 6;
    const int  tilesM  = M >> 4;
    const int  mt      = (int)(wid / ngroups);
    const int  ng      = (int)(wid % ngroups);
    if (mt >= tilesM) return;

    const int rsel = lane >> 4;     // 0: K 0..15 of chunk, 1: K 16..31
    const int lid  = lane & 15;     // row (A) / col (B) within 16-tile
    const int rowA = mt * 16 + lid;
    const int kseg = rsel * 16;

    vf8 acc0 = {}, acc1 = {}, acc2 = {}, acc3 = {};
    const float*    xrow = X  + (long)rowA * K + kseg;
    const _Float16* wrow = lw + (ng * 64 + lid) * KP + kseg;

    for (int k0 = 0; k0 < K; k0 += 32) {
        vh16 a = load_h16(xrow + k0);
        if (k0 + 32 < K) __builtin_prefetch(xrow + k0 + 32, 0, 1);  // global_prefetch_b8
        vh16 b0 = *reinterpret_cast<const vh16*>(wrow + k0);
        vh16 b1 = *reinterpret_cast<const vh16*>(wrow + k0 + 16 * KP);
        vh16 b2 = *reinterpret_cast<const vh16*>(wrow + k0 + 32 * KP);
        vh16 b3 = *reinterpret_cast<const vh16*>(wrow + k0 + 48 * KP);
        acc0 = __builtin_amdgcn_wmma_f32_16x16x32_f16(false, a, false, b0, (short)0, acc0, false, false);
        acc1 = __builtin_amdgcn_wmma_f32_16x16x32_f16(false, a, false, b1, (short)0, acc1, false, false);
        acc2 = __builtin_amdgcn_wmma_f32_16x16x32_f16(false, a, false, b2, (short)0, acc2, false, false);
        acc3 = __builtin_amdgcn_wmma_f32_16x16x32_f16(false, a, false, b3, (short)0, acc3, false, false);
    }

    // C/D layout: VGPR v holds M = v (+8 for lanes 16..31), N = lane&15.
    const int rbase = mt * 16 + rsel * 8;
#pragma unroll
    for (int t = 0; t < 4; ++t) {
        vf8 acc = (t == 0) ? acc0 : (t == 1) ? acc1 : (t == 2) ? acc2 : acc3;
        const int cidx = ng * 64 + t * 16 + lid;
        const float bb = bias ? bias[cidx] : 0.0f;
#pragma unroll
        for (int v = 0; v < 8; ++v) {
            const long r = rbase + v;
            float val = acc[v] + bb;
            if (res) val += res[r * (long)Nout + cidx];
            C[r * (long)Nout + cidx] = val;
        }
    }
}

// ---------------------------------------------------------------------------
// Elementwise / reduction kernels
// ---------------------------------------------------------------------------
__global__ void embed_k(const int* __restrict__ nt, const float* __restrict__ emb,
                        float* __restrict__ nh, int M) {
    long i = (long)blockIdx.x * blockDim.x + threadIdx.x;
    if (i >= (long)M * DD) return;
    int r = (int)(i >> 6), d = (int)(i & 63);
    nh[i] = emb[nt[r] * DD + d];
}

struct Mat3 { float m[9]; };

__global__ void edge_lin1_k(const float* __restrict__ ef, Mat3 R,
                            const float* __restrict__ w, const float* __restrict__ b,
                            float* __restrict__ out, int Ecnt) {
    long i = (long)blockIdx.x * blockDim.x + threadIdx.x;
    if (i >= (long)Ecnt * DD) return;
    int e = (int)(i >> 6), d = (int)(i & 63);
    float f0 = ef[e * 3 + 0], f1 = ef[e * 3 + 1], f2 = ef[e * 3 + 2];
    float t0 = f0 * R.m[0] + f1 * R.m[3] + f2 * R.m[6];
    float t1 = f0 * R.m[1] + f1 * R.m[4] + f2 * R.m[7];
    float t2 = f0 * R.m[2] + f1 * R.m[5] + f2 * R.m[8];
    out[i] = t0 * w[d * 3 + 0] + t1 * w[d * 3 + 1] + t2 * w[d * 3 + 2] + b[d];
}

__global__ void fill_k(float* __restrict__ p, float v, long n) {
    long i = (long)blockIdx.x * blockDim.x + threadIdx.x;
    if (i < n) p[i] = v;
}

// Batchnorm stats: per-block LDS partials, one atomic per feature per block.
__global__ void bn_stats_k(const float* __restrict__ X, int M, int D, float* __restrict__ st) {
    __shared__ float sm1[256], sm2[256];
    const int t = threadIdx.x;
    const int d = t & (D - 1);
    const int g = t / D;
    const int ngroups = blockDim.x / D;
    const int r0 = blockIdx.x * 128;
    float s1 = 0.f, s2 = 0.f;
    for (int rr = g; rr < 128; rr += ngroups) {
        int r = r0 + rr;
        if (r >= M) break;
        float x = X[(long)r * D + d];
        s1 += x; s2 += x * x;
    }
    sm1[t] = s1; sm2[t] = s2;
    __syncthreads();
    if (t < D) {
        for (int gg = 1; gg < ngroups; ++gg) { s1 += sm1[gg * D + t]; s2 += sm2[gg * D + t]; }
        atomicAdd(&st[t], s1);
        atomicAdd(&st[D + t], s2);
    }
}

__global__ void bn_finalize_k(float* __restrict__ st, int D, float invM) {
    int d = threadIdx.x;
    if (d < D) {
        float mean = st[d] * invM;
        float var  = st[D + d] * invM - mean * mean;
        st[d]     = mean;
        st[D + d] = rsqrtf(var + 1e-5f);
    }
}

// act: 0 none, 1 relu, 3 prelu(alpha).  accum: Y += result.
__global__ void bn_apply_k(const float* __restrict__ X, float* __restrict__ Y,
                           const float* __restrict__ st,
                           const float* __restrict__ g, const float* __restrict__ b,
                           long total, int D, int act, const float* __restrict__ alpha,
                           int accum) {
    long i0 = (long)blockIdx.x * blockDim.x + threadIdx.x;
    long stride = (long)gridDim.x * blockDim.x;
    for (long i = i0; i < total; i += stride) {
        int d = (int)(i % D);
        float v = (X[i] - st[d]) * st[D + d] * g[d] + b[d];
        if (act == 1) v = v > 0.f ? v : 0.f;
        else if (act == 3) { float a = alpha[0]; v = v >= 0.f ? v : a * v; }
        if (accum) Y[i] += v; else Y[i] = v;
    }
}

// f = G + f_ni[src] + f_nj[dst] + bias, leaky-relu, per-head attention score,
// atomic segment-max into m[dst].  blockDim == HD, LDS reduce per 64-thr head.
__global__ void edge_combine_k(float* __restrict__ F,
                               const float* __restrict__ fni, const float* __restrict__ fnj,
                               const float* __restrict__ ebias, const float* __restrict__ attn,
                               const int* __restrict__ src, const int* __restrict__ dst,
                               float* __restrict__ scores, float* __restrict__ mbuf,
                               int HD, int H) {
    extern __shared__ float sm[];
    const int e = blockIdx.x;
    const int t = threadIdx.x;
    const int s = src[e], d = dst[e];
    const int h = t >> 6;
    float v = F[(long)e * HD + t] + fni[(long)s * HD + t] + fnj[(long)d * HD + t] + ebias[t];
    v = v >= 0.f ? v : 0.01f * v;
    F[(long)e * HD + t] = v;
    sm[t] = v * attn[(h << 6) + (t & 63)];
    __syncthreads();
    for (int stp = 32; stp > 0; stp >>= 1) {
        if ((t & 63) < stp) sm[t] += sm[t + stp];
        __syncthreads();
    }
    if ((t & 63) == 0) {
        float sc = sm[t];
        scores[(long)e * H + h] = sc;
        atomicMaxF(&mbuf[(long)d * H + h], sc);
    }
}

__global__ void edge_softmax_k(const float* __restrict__ scores, const float* __restrict__ mbuf,
                               const int* __restrict__ dst, float* __restrict__ exs,
                               float* __restrict__ ssum, int Ecnt, int H) {
    long i = (long)blockIdx.x * blockDim.x + threadIdx.x;
    if (i >= (long)Ecnt * H) return;
    int e = (int)(i / H), h = (int)(i % H);
    float ex = __expf(scores[i] - mbuf[(long)dst[e] * H + h]);
    exs[i] = ex;
    atomicAdd(&ssum[(long)dst[e] * H + h], ex);
}

__global__ void edge_agg_k(const float* __restrict__ hsv, const float* __restrict__ exs,
                           const float* __restrict__ ssum,
                           const int* __restrict__ src, const int* __restrict__ dst,
                           float* __restrict__ nout, int HD, int H) {
    const int e = blockIdx.x;
    const int t = threadIdx.x;
    const int h = t >> 6;
    const int s = src[e], d = dst[e];
    float a = exs[(long)e * H + h] / (ssum[(long)d * H + h] + 1e-16f);
    atomicAdd(&nout[(long)d * HD + t], hsv[(long)s * HD + t] * a);
}

__global__ void dec_out_k(const float* __restrict__ X, const float* __restrict__ w,
                          const float* __restrict__ b, float* __restrict__ out, int M) {
    int r = blockIdx.x * blockDim.x + threadIdx.x;
    if (r >= M) return;
    float s = 0.f;
#pragma unroll
    for (int k = 0; k < DD; ++k) s += X[(long)r * DD + k] * w[k];
    out[r] = s + b[0];
}

// ---------------------------------------------------------------------------
// Host side
// ---------------------------------------------------------------------------
static void make_rot(int i, float* R) {
    float c = cosf((float)i), s = sinf((float)i);
    float Rt[9] = {c, -s, 0, s, c, 0, 0, 0, 1};
    float Rp[9] = {c, 0, -s, 0, 1, 0, s, 0, c};
    float Rg[9] = {1, 0, 0, 0, c, -s, 0, s, c};
    float T[9];
    for (int r = 0; r < 3; r++) for (int q = 0; q < 3; q++) {
        float a = 0; for (int k = 0; k < 3; k++) a += Rt[r * 3 + k] * Rp[k * 3 + q];
        T[r * 3 + q] = a;
    }
    for (int r = 0; r < 3; r++) for (int q = 0; q < 3; q++) {
        float a = 0; for (int k = 0; k < 3; k++) a += T[r * 3 + k] * Rg[k * 3 + q];
        R[r * 3 + q] = a;
    }
}

static int ilog2(int v) { int b = 0; while ((1 << b) < v) ++b; return b; }

extern "C" void kernel_launch(void* const* d_in, const int* in_sizes, int n_in,
                              void* d_out, int out_size, void* d_ws, size_t ws_size,
                              hipStream_t stream) {
    (void)n_in; (void)out_size; (void)ws_size;
    const int N = in_sizes[0];      // nfeats: N x 1
    const int E = in_sizes[2];      // src: E
    const int HEADS[3] = {4, 4, 1};

    const int*   nfeats = (const int*)d_in[0];
    const float* efeats = (const float*)d_in[1];
    const int*   src    = (const int*)d_in[2];
    const int*   dst    = (const int*)d_in[3];

    int ix = 4;
    auto nextf = [&]() -> const float* { return (const float*)d_in[ix++]; };
    auto nextlin = [&](const float*& w, const float*& b) {
        const float* p0 = (const float*)d_in[ix];     int s0 = in_sizes[ix];
        const float* p1 = (const float*)d_in[ix + 1]; int s1 = in_sizes[ix + 1];
        ix += 2;
        if (s0 >= s1) { w = p0; b = p1; } else { w = p1; b = p0; }
    };

    const float* embed = nextf();
    const float *enc1g = nextf(), *enc1b = nextf();
    const float *encw, *encb;  nextlin(encw, encb);
    const float *enc2g = nextf(), *enc2b = nextf();
    const float *ee1w, *ee1b;  nextlin(ee1w, ee1b);
    const float *eb1g = nextf(), *eb1b = nextf();
    const float *ee2w, *ee2b;  nextlin(ee2w, ee2b);
    const float *eb2g = nextf(), *eb2b = nextf();

    struct LayerP { const float *fcw,*fcb,*wni,*wfij,*wnj,*attn,*ebias,*nbg,*nbb,*ebg,*ebb,*pnw,*pnb,*pew,*peb; };
    LayerP LP[3];
    for (int l = 0; l < 3; l++) {
        nextlin(LP[l].fcw, LP[l].fcb);
        LP[l].wni  = nextf();
        LP[l].wfij = nextf();
        LP[l].wnj  = nextf();
        LP[l].attn = nextf();
        LP[l].ebias= nextf();
        LP[l].nbg = nextf(); LP[l].nbb = nextf();
        LP[l].ebg = nextf(); LP[l].ebb = nextf();
        nextlin(LP[l].pnw, LP[l].pnb);
        nextlin(LP[l].pew, LP[l].peb);
    }
    struct DecP { const float *w,*b,*g,*bb,*al; };
    DecP DP[4];
    for (int k = 0; k < 4; k++) {
        nextlin(DP[k].w, DP[k].b);
        DP[k].g = nextf(); DP[k].bb = nextf(); DP[k].al = nextf();
    }
    const float *dow, *dob; nextlin(dow, dob);

    // ---- workspace bump allocator ----
    char* wsp = (char*)d_ws;
    size_t off = 0;
    auto alloc = [&](size_t nfloats) -> float* {
        float* p = (float*)(wsp + off);
        off += ((nfloats * sizeof(float) + 255) & ~(size_t)255);
        return p;
    };
    float* nhA   = alloc((size_t)N * 64);
    float* nhB   = alloc((size_t)N * 64);
    float* eh    = alloc((size_t)E * 64);
    float* e1    = alloc((size_t)E * 64);
    float* e2    = alloc((size_t)E * 64);
    float* fbuf  = alloc((size_t)E * 256);
    float* fni   = alloc((size_t)N * 256);
    float* fnj   = alloc((size_t)N * 256);
    float* hsv   = alloc((size_t)N * 256);
    float* nout  = alloc((size_t)N * 256);
    float* scores= alloc((size_t)E * 4);
    float* exs   = alloc((size_t)E * 4);
    float* mbuf  = alloc((size_t)N * 4);
    float* ssum  = alloc((size_t)N * 4);
    float* stats = alloc(512);

    float* outp   = (float*)d_out;
    float* hlist[3] = { outp, outp + (long)N * 64, outp + 2L * N * 64 };
    float* hfinal = outp + 3L * N * 64;

    auto cdiv = [](long a, long b) -> int { return (int)((a + b - 1) / b); };

    auto gemm = [&](const float* X, const float* W, const float* bias, const float* res,
                    float* C, int M, int K, int Nout) {
        long waves = (long)(M / 16) * (Nout / 64);
        size_t ldsB = (size_t)Nout * (K + 8) * sizeof(_Float16);
        gemm_nt_wmma<<<dim3(cdiv(waves, 8)), dim3(256), ldsB, stream>>>(
            X, W, bias, res, C, M, K, Nout, ilog2(K));
    };
    auto bn = [&](const float* X, float* Y, const float* g, const float* b,
                  int M, int D, int act, const float* alpha, int accum) {
        hipMemsetAsync(stats, 0, 2 * D * sizeof(float), stream);
        bn_stats_k<<<dim3(cdiv(M, 128)), dim3(256), 0, stream>>>(X, M, D, stats);
        bn_finalize_k<<<dim3(1), dim3(D), 0, stream>>>(stats, D, 1.0f / (float)M);
        long total = (long)M * D;
        int blocks = (int)((total + 255) / 256); if (blocks > 4096) blocks = 4096;
        bn_apply_k<<<dim3(blocks), dim3(256), 0, stream>>>(X, Y, stats, g, b, total, D, act, alpha, accum);
    };

    // ---------------- node encoder ----------------
    embed_k<<<dim3(cdiv((long)N * 64, 256)), dim3(256), 0, stream>>>(nfeats, embed, nhA, N);
    bn(nhA, nhA, enc1g, enc1b, N, 64, 1, nullptr, 0);
    gemm(nhA, encw, encb, nullptr, nhB, N, 64, 64);
    bn(nhB, nhB, enc2g, enc2b, N, 64, 1, nullptr, 0);
    float* nh = nhB;

    // ---------------- edge encoder (2 rotations, summed) ----------------
    hipMemsetAsync(eh, 0, (size_t)E * 64 * sizeof(float), stream);
    for (int it = 0; it < 2; ++it) {
        Mat3 R; make_rot(it, R.m);
        edge_lin1_k<<<dim3(cdiv((long)E * 64, 256)), dim3(256), 0, stream>>>(efeats, R, ee1w, ee1b, e1, E);
        bn(e1, e1, eb1g, eb1b, E, 64, 1, nullptr, 0);
        gemm(e1, ee2w, ee2b, nullptr, e2, E, 64, 64);
        bn(e2, eh, eb2g, eb2b, E, 64, 1, nullptr, 1);   // eh += relu(bn(x))
    }
    float* ehCur = eh; float* eSp1 = e1; float* eSp2 = e2;

    // ---------------- EGAT layers ----------------
    for (int li = 0; li < 3; ++li) {
        const LayerP& L = LP[li];
        const int H = HEADS[li], HD = H * 64;

        gemm(nh,    L.wni,  nullptr, nullptr, fni,  N, 64, HD);
        gemm(nh,    L.wnj,  nullptr, nullptr, fnj,  N, 64, HD);
        gemm(nh,    L.fcw,  L.fcb,   nullptr, hsv,  N, 64, HD);
        gemm(ehCur, L.wfij, nullptr, nullptr, fbuf, E, 64, HD);

        fill_k<<<dim3(cdiv((long)N * H, 256)), dim3(256), 0, stream>>>(mbuf, -INFINITY, (long)N * H);
        hipMemsetAsync(ssum, 0, (size_t)N * H * sizeof(float), stream);
        hipMemsetAsync(nout, 0, (size_t)N * HD * sizeof(float), stream);

        edge_combine_k<<<dim3(E), dim3(HD), HD * sizeof(float), stream>>>(
            fbuf, fni, fnj, L.ebias, L.attn, src, dst, scores, mbuf, HD, H);
        edge_softmax_k<<<dim3(cdiv((long)E * H, 256)), dim3(256), 0, stream>>>(
            scores, mbuf, dst, exs, ssum, E, H);
        edge_agg_k<<<dim3(E), dim3(HD), 0, stream>>>(hsv, exs, ssum, src, dst, nout, HD, H);

        bn(nout, nout, L.nbg, L.nbb, N, HD, 0, nullptr, 0);
        bn(fbuf, fbuf, L.ebg, L.ebb, E, HD, 0, nullptr, 0);

        gemm(nout, L.pnw, L.pnb, nh,    hlist[li], N, HD, 64);   // + node residual
        gemm(fbuf, L.pew, L.peb, ehCur, eSp1,      E, HD, 64);   // + edge residual
        nh = hlist[li];
        float* old = ehCur; ehCur = eSp1; eSp1 = eSp2; eSp2 = old;
    }

    // ---------------- decoder ----------------
    const float* hc = nh;           // hlist[2]
    float* dA = nhA; float* dB = nhB;
    for (int k = 0; k < 4; ++k) {
        gemm(hc, DP[k].w, DP[k].b, nullptr, dA, N, 64, 64);
        bn(dA, dA, DP[k].g, DP[k].bb, N, 64, 3, DP[k].al, 0);
        hc = dA; float* t = dA; dA = dB; dB = t;
    }
    dec_out_k<<<dim3(cdiv(N, 256)), dim3(256), 0, stream>>>(hc, dow, dob, hfinal, N);
}
